// KuramotoOscillatorNetwork_26379689132369
// MI455X (gfx1250) — matrix operand order repaired and checked by reading, not compiled, not run on hardware
//
#include <hip/hip_runtime.h>

typedef __attribute__((ext_vector_type(16))) __bf16 v16bf;
typedef __attribute__((ext_vector_type(8)))  __bf16 v8bf;
typedef __attribute__((ext_vector_type(8)))  float  v8f;

#define BSZ 256
#define DSZ 256
#define NSZ 16
#define KK  4096      // D*N  (contraction length and output column count)
#define TSTEPS 25
#define GAMMA 0.1f
#define ALPHA 1.0f
#define NITER (KK / 32)

__device__ __forceinline__ v8f wmma_bf16(v16bf a, v16bf b, v8f c) {
    return __builtin_amdgcn_wmma_f32_16x16x32_bf16(
        /*neg_a=*/false, a, /*neg_b=*/false, b,
        /*c_mod=*/(short)0, c, /*reuse_a=*/false, /*reuse_b=*/false);
}

// 512B async copy: 32 lanes x 16B, global -> LDS (tracked by ASYNCcnt)
__device__ __forceinline__ void async_b128(uint32_t lds_off, uint64_t gaddr) {
    asm volatile("global_load_async_to_lds_b128 %0, %1, off"
                 :: "v"(lds_off), "v"(gaddr) : "memory");
}

// ---------------------------------------------------------------------------
// One-time: J[i,j,k,l] (fp32) -> Jt[n=(i*16+k)][kk=(j*16+l)] (bf16)
// Output row n is contiguous in kk -> B-operand loads become contiguous.
// ---------------------------------------------------------------------------
__global__ void convertJ(const float* __restrict__ J, __bf16* __restrict__ Jt) {
    int tid = blockIdx.x * blockDim.x + threadIdx.x;   // 4096 * 256 threads
    int n = tid >> 8;           // (i,k) column index
    int j = tid & 255;
    int i = n >> 4, k = n & 15;
    const float* src = J + ((size_t)((i * DSZ + j) * NSZ + k)) * NSZ; // 16 contig (l)
    __bf16* dst = Jt + (size_t)n * KK + j * 16;
    #pragma unroll
    for (int l = 0; l < 16; ++l) dst[l] = (__bf16)src[l];
}

// ---------------------------------------------------------------------------
// Init: x = normalize(noise + ALPHA * (emb[b,d] * W_d[d,:] + b_d[d,:]))
// ---------------------------------------------------------------------------
__global__ void initX(const float* __restrict__ emb1, const float* __restrict__ emb2,
                      const float* __restrict__ Wd,   const float* __restrict__ bd,
                      const float* __restrict__ noi1, const float* __restrict__ noi2,
                      float* __restrict__ xf, __bf16* __restrict__ xb) {
    int tid = blockIdx.x * blockDim.x + threadIdx.x;   // 2*B*D = 131072
    int s = tid >> 16;
    int rem = tid & 65535;
    int b = rem >> 8, d = rem & 255;
    const float* emb = s ? emb2 : emb1;
    const float* noi = s ? noi2 : noi1;
    float e = emb[b * DSZ + d];
    size_t off = (size_t)tid * NSZ;
    float x[NSZ]; float ss = 0.f;
    size_t noff = ((size_t)b * DSZ + d) * NSZ;
    #pragma unroll
    for (int n = 0; n < NSZ; ++n) {
        float c = e * Wd[d * NSZ + n] + bd[d * NSZ + n];
        float v = noi[noff + n] + ALPHA * c;
        x[n] = v; ss += v * v;
    }
    float inv = 1.f / fmaxf(sqrtf(ss), 1e-12f);
    #pragma unroll
    for (int n = 0; n < NSZ; ++n) {
        float v = x[n] * inv;
        xf[off + n] = v;
        xb[off + n] = (__bf16)v;
    }
}

// ---------------------------------------------------------------------------
// Fused coupling GEMM with async->LDS staged B operands:
//   F[0][m][n] = x1[m,:]·JtIn[n,:] + x2[m,:]·JtOut[n,:]
//   F[1][m][n] = x2[m,:]·JtIn[n,:] + x1[m,:]·JtOut[n,:]
// Block = 4 waves covering 64 rows x 64 cols; all waves share the same four
// 16-col J tile-pairs, staged triple-buffered in LDS via async loads.
// Wave w stages tile t=w (Bin+Bout = 4 x 512B async b128) per K-step.
// grid = (64, 4), block = 128.
// ---------------------------------------------------------------------------
__global__ __launch_bounds__(128) void gemmF(const __bf16* __restrict__ xb,
                                             const __bf16* __restrict__ JtIn,
                                             const __bf16* __restrict__ JtOut,
                                             float* __restrict__ F) {
    __shared__ __align__(32) char smem[3 * 8192];   // 3 stages x (4 tiles x 2 mats x 1KB)

    const int lane = threadIdx.x & 31;
    const int wave = threadIdx.x >> 5;
    const int m0 = blockIdx.y * 64 + wave * 16;     // this wave's row tile
    const int n0 = blockIdx.x * 64;                 // block's 64-col strip
    const int lr = lane & 15;                       // row-within-tile (M for A, N for B)
    const int hk = lane >> 4;                       // lane-half selector

    // ---- async staging setup: wave w copies B tile t=w (rows n0+w*16..+15) ----
    const uint32_t lrow  = (uint32_t)(lane >> 2);        // 0..7 (chunk row)
    const uint32_t lboff = (uint32_t)((lane & 3) * 16);  // byte offset in 64B row
    const uint64_t gIn  = (uint64_t)(uintptr_t)JtIn  + (uint64_t)(n0 + wave * 16) * (KK * 2);
    const uint64_t gOut = (uint64_t)(uintptr_t)JtOut + (uint64_t)(n0 + wave * 16) * (KK * 2);
    const uint32_t ldsBase = (uint32_t)(uintptr_t)smem;

    #define ISSUE_STAGE(st)                                                          \
    do {                                                                             \
        uint32_t sb_  = ldsBase + (uint32_t)((st) % 3) * 8192;                       \
        uint64_t kb_  = (uint64_t)(st) * 64;                                         \
        uint32_t din_ = sb_ + (uint32_t)(wave * 2 + 0) * 1024;                       \
        uint32_t dou_ = sb_ + (uint32_t)(wave * 2 + 1) * 1024;                       \
        async_b128(din_ + lrow * 64 + lboff,        gIn  + lrow * 8192ull + kb_ + lboff); \
        async_b128(din_ + (lrow + 8) * 64 + lboff,  gIn  + (lrow + 8) * 8192ull + kb_ + lboff); \
        async_b128(dou_ + lrow * 64 + lboff,        gOut + lrow * 8192ull + kb_ + lboff); \
        async_b128(dou_ + (lrow + 8) * 64 + lboff,  gOut + (lrow + 8) * 8192ull + kb_ + lboff); \
    } while (0)

    // A operand (this code's contiguous k order, same as B): per lane two b128 chunks:
    //   lanes 0-15 : K = [0..7]  and [8..15]
    //   lanes 16-31: K = [16..23] and [24..31]
    const __bf16* a1p = xb + (size_t)(m0 + lr) * KK + hk * 16;
    const __bf16* a2p = a1p + (size_t)BSZ * KK;

    v8f acc[2][4];
    #pragma unroll
    for (int s = 0; s < 2; ++s)
        #pragma unroll
        for (int t = 0; t < 4; ++t) acc[s][t] = (v8f){};

    // prologue: stages 0 and 1 in flight (8 outstanding async ops)
    ISSUE_STAGE(0);
    ISSUE_STAGE(1);

    for (int i = 0; i < NITER; ++i) {
        // retire my stage-i copies (in-order; stage i+1's 4 may stay in flight)
        if (i + 1 < NITER) asm volatile("s_wait_asynccnt 0x4" ::: "memory");
        else               asm volatile("s_wait_asynccnt 0x0" ::: "memory");
        __syncthreads();              // stage i visible block-wide; compute(i-1) done everywhere
        if (i + 2 < NITER) ISSUE_STAGE(i + 2);   // overwrites buffer last read at iter i-1

        const int k = i * 32;
        v8bf a1lo = *(const v8bf*)(a1p + k);
        v8bf a1hi = *(const v8bf*)(a1p + k + 8);
        v8bf a2lo = *(const v8bf*)(a2p + k);
        v8bf a2hi = *(const v8bf*)(a2p + k + 8);
        v16bf a1 = __builtin_shufflevector(a1lo, a1hi, 0,1,2,3,4,5,6,7,8,9,10,11,12,13,14,15);
        v16bf a2 = __builtin_shufflevector(a2lo, a2hi, 0,1,2,3,4,5,6,7,8,9,10,11,12,13,14,15);

        // B operands from LDS: lane = column lr, contiguous 16 halves at K half hk*16
        const char* sp = smem + (i % 3) * 8192 + lr * 64 + hk * 32;
        #pragma unroll
        for (int t = 0; t < 4; ++t) {
            v16bf bi = *(const v16bf*)(sp + (t * 2 + 0) * 1024);
            v16bf bo = *(const v16bf*)(sp + (t * 2 + 1) * 1024);
            acc[0][t] = wmma_bf16(a1, bi, acc[0][t]);
            acc[0][t] = wmma_bf16(a2, bo, acc[0][t]);
            acc[1][t] = wmma_bf16(a2, bi, acc[1][t]);
            acc[1][t] = wmma_bf16(a1, bo, acc[1][t]);
        }
    }
    #undef ISSUE_STAGE

    // C/D layout: VGPR r -> (M = r + 8*hk, N = lr)
    const int mrow = m0 + hk * 8;
    #pragma unroll
    for (int s = 0; s < 2; ++s)
        #pragma unroll
        for (int t = 0; t < 4; ++t)
            #pragma unroll
            for (int r = 0; r < 8; ++r)
                F[(size_t)s * BSZ * KK + (size_t)(mrow + r) * KK + n0 + t * 16 + lr] = acc[s][t][r];
}

// ---------------------------------------------------------------------------
// Per-step state update (fp32): f = F + c ; p = f - (x.f)x ; om = Omega x ;
// x <- normalize(x + GAMMA*(om + p)).  thread = (s, b, d), in-place on x.
// ---------------------------------------------------------------------------
__global__ void updateX(const float* __restrict__ emb1, const float* __restrict__ emb2,
                        const float* __restrict__ Wd,   const float* __restrict__ bd,
                        const float* __restrict__ Omega,
                        const float* __restrict__ F,
                        float* __restrict__ xf, __bf16* __restrict__ xb) {
    int tid = blockIdx.x * blockDim.x + threadIdx.x;   // 2*B*D
    int s = tid >> 16;
    int rem = tid & 65535;
    int b = rem >> 8, d = rem & 255;
    size_t off = (size_t)tid * NSZ;

    float e = (s ? emb2 : emb1)[b * DSZ + d];
    float x[NSZ], f[NSZ];
    float dot = 0.f;
    #pragma unroll
    for (int n = 0; n < NSZ; ++n) {
        x[n] = xf[off + n];
        f[n] = F[off + n] + e * Wd[d * NSZ + n] + bd[d * NSZ + n];
        dot += x[n] * f[n];
    }
    const float* Om = Omega + (size_t)d * NSZ * NSZ;
    float y[NSZ]; float ss = 0.f;
    #pragma unroll
    for (int n = 0; n < NSZ; ++n) {
        float om = 0.f;
        #pragma unroll
        for (int m = 0; m < NSZ; ++m) om += Om[n * NSZ + m] * x[m];
        float v = x[n] + GAMMA * (om + (f[n] - dot * x[n]));
        y[n] = v; ss += v * v;
    }
    float inv = 1.f / fmaxf(sqrtf(ss), 1e-12f);
    #pragma unroll
    for (int n = 0; n < NSZ; ++n) {
        float v = y[n] * inv;
        xf[off + n] = v;
        xb[off + n] = (__bf16)v;
    }
}

// ---------------------------------------------------------------------------
extern "C" void kernel_launch(void* const* d_in, const int* in_sizes, int n_in,
                              void* d_out, int out_size, void* d_ws, size_t ws_size,
                              hipStream_t stream) {
    const float* emb1  = (const float*)d_in[0];
    const float* emb2  = (const float*)d_in[1];
    const float* Wd    = (const float*)d_in[2];
    const float* bd    = (const float*)d_in[3];
    const float* Jin   = (const float*)d_in[4];
    const float* Jout  = (const float*)d_in[5];
    const float* Omega = (const float*)d_in[6];
    const float* noi1  = (const float*)d_in[7];
    const float* noi2  = (const float*)d_in[8];

    // workspace carve-up (all offsets 256B aligned)
    char* ws = (char*)d_ws;
    const size_t szJ  = (size_t)KK * KK * sizeof(__bf16);       // 32 MB each
    const size_t szXF = (size_t)2 * BSZ * KK * sizeof(float);   // 8 MB
    const size_t szXB = (size_t)2 * BSZ * KK * sizeof(__bf16);  // 4 MB
    __bf16* JtIn  = (__bf16*)ws;
    __bf16* JtOut = (__bf16*)(ws + szJ);
    float*  xf    = (float*)(ws + 2 * szJ);
    __bf16* xbuf  = (__bf16*)(ws + 2 * szJ + szXF);
    float*  F     = (float*)(ws + 2 * szJ + szXF + szXB);

    // 1) one-time bf16 transpose of both coupling tensors
    convertJ<<<4096, 256, 0, stream>>>(Jin,  JtIn);
    convertJ<<<4096, 256, 0, stream>>>(Jout, JtOut);

    // 2) oscillator init on the sphere
    initX<<<512, 256, 0, stream>>>(emb1, emb2, Wd, bd, noi1, noi2, xf, xbuf);

    // 3) T coupled sphere-flow steps: async-LDS staged bf16 WMMA GEMM + fp32 update
    for (int t = 0; t < TSTEPS; ++t) {
        gemmF<<<dim3(64, 4), 128, 0, stream>>>(xbuf, JtIn, JtOut, F);
        updateX<<<512, 256, 0, stream>>>(emb1, emb2, Wd, bd, Omega, F, xf, xbuf);
    }

    // 4) output = stack([x1, x2]) : already laid out [2, B, D, N] fp32
    hipMemcpyAsync(d_out, xf, (size_t)2 * BSZ * DSZ * NSZ * sizeof(float),
                   hipMemcpyDeviceToDevice, stream);
}
